// TEA_4114578669900
// MI455X (gfx1250) — matrix operand order repaired and checked by reading, not compiled, not run on hardware
//
#include <hip/hip_runtime.h>

typedef unsigned short u16;
typedef __attribute__((ext_vector_type(16))) __bf16 v16bf;
typedef __attribute__((ext_vector_type(8)))  float  v8f;

union Frag { v16bf v; uint4 q[2]; };

__device__ __forceinline__ u16 f2bf(float f) {
    unsigned int u = __float_as_uint(f);
    unsigned int r = (u + 0x7FFFu + ((u >> 16) & 1u)) >> 16;
    return (u16)r;
}

// ---------------------------------------------------------------------------
// CDNA5 async / TDM data movement
// ---------------------------------------------------------------------------
#define ENABLE_TDM 1

#if __has_builtin(__builtin_amdgcn_global_load_async_to_lds_b128)
#define HAVE_ASYNC 1
typedef int v4i32 __attribute__((vector_size(16)));
typedef __attribute__((address_space(1))) v4i32 gas_v4i32;
typedef __attribute__((address_space(3))) v4i32 las_v4i32;
__device__ __forceinline__ void async_cp_b128(const u16* g, u16* l) {
    __builtin_amdgcn_global_load_async_to_lds_b128(
        (gas_v4i32*)(unsigned long long)g, (las_v4i32*)l, 0, 0);
}
#define CP_B128(g, l) async_cp_b128((g), (l))
#if __has_builtin(__builtin_amdgcn_s_wait_asynccnt)
#define WAIT_ASYNC() __builtin_amdgcn_s_wait_asynccnt(0)
#else
#define WAIT_ASYNC() asm volatile("s_wait_asynccnt 0x0" ::: "memory")
#endif
#else
#define HAVE_ASYNC 0
#define CP_B128(g, l) (*(uint4*)(l) = *(const uint4*)(g))
#define WAIT_ASYNC() ((void)0)
#endif

#if ENABLE_TDM
typedef __attribute__((ext_vector_type(4))) unsigned tdm_g0_t;
typedef __attribute__((ext_vector_type(8))) unsigned tdm_g1_t;

__device__ __forceinline__ unsigned lds_off_of(const void* p) {
    return (unsigned)(unsigned long long)
        (__attribute__((address_space(3))) const void*)p;
}
// 2D bf16 tile DMA: tile (tile_d1 rows x tile_d0 elems) from row-major tensor
// (row length stride_d0 elems, bounds tensor_d0 x tensor_d1) into contiguous LDS.
__device__ __forceinline__ void tdm_issue_2d_bf16(
    unsigned lds_byte_off, const void* gaddr, unsigned tensor_d0,
    unsigned tensor_d1, unsigned tile_d0, unsigned tile_d1, unsigned stride_d0) {
    unsigned long long ga = (unsigned long long)gaddr;
    tdm_g0_t g0;
    g0[0] = 1u;                                   // count=1 valid descriptor
    g0[1] = lds_byte_off;                         // lds_addr
    g0[2] = (unsigned)ga;                         // global_addr[31:0]
    g0[3] = ((unsigned)(ga >> 32) & 0x01FFFFFFu)  // global_addr[56:32]
            | 0x80000000u;                        // type=2 ("image")
    tdm_g1_t g1;
    g1[0] = 0x10000u;                             // data_size=1 -> 2 bytes
    g1[1] = (tensor_d0 & 0xFFFFu) << 16;          // tensor_dim0[15:0]
    g1[2] = (tensor_d0 >> 16) | ((tensor_d1 & 0xFFFFu) << 16);
    g1[3] = (tensor_d1 >> 16) | (tile_d0 << 16);  // tile_dim0
    g1[4] = tile_d1 & 0xFFFFu;                    // tile_dim1 (tile_dim2=0)
    g1[5] = stride_d0;                            // tensor_dim0_stride[31:0]
    g1[6] = 0u;
    g1[7] = 0u;
    asm volatile("tensor_load_to_lds %0, %1" :: "s"(g0), "s"(g1) : "memory");
}
__device__ __forceinline__ void tdm_wait() {
    asm volatile("s_wait_tensorcnt 0x0" ::: "memory");
}
#endif

// ---------------------------------------------------------------------------
// conversion / permutation helpers
// ---------------------------------------------------------------------------
__global__ void cvt_bf_kernel(const float* __restrict__ s, u16* __restrict__ d, size_t n) {
    size_t i = (size_t)blockIdx.x * 256 + threadIdx.x;
    if (i < n) d[i] = f2bf(s[i]);
}

// OIHW -> O,(kh*KW+kw),I  bf16
__global__ void cvt_conv_w(const float* __restrict__ src, u16* __restrict__ dst,
                           int O, int Ci, int Ks) {
    int idx = blockIdx.x * 256 + threadIdx.x;
    int total = O * Ci * Ks * Ks;
    if (idx >= total) return;
    int ci = idx % Ci; int t = idx / Ci;
    int s = t % (Ks * Ks); int o = t / (Ks * Ks);
    int kh = s / Ks, kw = s % Ks;
    dst[idx] = f2bf(src[(((size_t)o * Ci + ci) * Ks + kh) * Ks + kw]);
}

// enc_w [256][ci*64 + p] -> [256][p*256 + ci]  (p = oh*8+ow), matches NHWC h4 flatten
__global__ void cvt_enc_w(const float* __restrict__ src, u16* __restrict__ dst) {
    int idx = blockIdx.x * 256 + threadIdx.x;
    if (idx >= 256 * 16384) return;
    int n = idx / 16384; int r = idx % 16384;
    int p = r / 256; int ci = r % 256;
    dst[idx] = f2bf(src[(size_t)n * 16384 + ci * 64 + p]);
}

__global__ void row_sqnorm(const float* __restrict__ src, float* __restrict__ dst,
                           int R, int Dd) {
    int r = blockIdx.x * 256 + threadIdx.x;
    if (r >= R) return;
    const float* p = src + (size_t)r * Dd;
    float s = 0.f;
    for (int i = 0; i < Dd; i++) { float v = p[i]; s += v * v; }
    dst[r] = s;
}

// ---------------------------------------------------------------------------
// conv1: direct f32 (3 input channels, K=75 — not worth WMMA), relu, NHWC bf16 out
// ---------------------------------------------------------------------------
__global__ void conv1_kernel(const float* __restrict__ x, const float* __restrict__ w1,
                             const float* __restrict__ b1, u16* __restrict__ out) {
    int idx = blockIdx.x * 256 + threadIdx.x;
    const int total = 4096 * 15 * 15 * 32;
    if (idx >= total) return;
    int co = idx & 31; int p = idx >> 5;
    int ow = p % 15; p /= 15; int oh = p % 15; int b = p / 15;
    float acc = b1[co];
    for (int ci = 0; ci < 3; ci++)
        for (int kh = 0; kh < 5; kh++) {
            int ih = oh * 2 - 1 + kh;
            if (ih < 0 || ih >= 32) continue;
            for (int kw = 0; kw < 5; kw++) {
                int iw = ow * 2 - 1 + kw;
                if (iw < 0 || iw >= 32) continue;
                acc += x[(((size_t)b * 3 + ci) * 32 + ih) * 32 + iw] *
                       w1[((co * 3 + ci) * 5 + kh) * 5 + kw];
            }
        }
    out[idx] = f2bf(fmaxf(acc, 0.f));
}

// ---------------------------------------------------------------------------
// implicit-GEMM conv: NHWC bf16 act, O,(HW),I bf16 weights, WMMA bf16, relu,
// NHWC bf16 out.  Double-buffered LDS; activations via async global->LDS,
// weight tiles via TDM tensor_load_to_lds (wave 0).  Cin % 32 == 0.
// ---------------------------------------------------------------------------
template <int Cin, int KS, int STRIDE, int PAD, int Hin, int Win,
          int Hout, int Wout, int Cout, int TN>
__global__ __launch_bounds__(256) void conv_igemm(
    const u16* __restrict__ act, const u16* __restrict__ wt,
    const float* __restrict__ bias, u16* __restrict__ out) {
    constexpr int K  = KS * KS * Cin;
    constexpr int NT = TN / 16;
    __shared__ u16 As[2][128 * 32];
    __shared__ u16 Bs[2][TN * 32];
    int tid = threadIdx.x;
    int wv = tid >> 5, lane = tid & 31;
    int lh = lane >> 4, ln = lane & 15;
    int n0 = blockIdx.y * TN;
    int mloc = tid >> 1, half = tid & 1;
    int m = blockIdx.x * 128 + mloc;
    int bimg = m / (Hout * Wout); int rem = m % (Hout * Wout);
    int oh = rem / Wout, ow = rem % Wout;

    v8f acc[NT];
    for (int i = 0; i < NT; i++)
        for (int r = 0; r < 8; r++) acc[i][r] = 0.f;

    auto loadA = [&](int k0, int buf) {
        int s = k0 / Cin;
        int kh = s / KS, kw = s % KS;
        int ih = oh * STRIDE - PAD + kh;
        int iw = ow * STRIDE - PAD + kw;
        int ci = (k0 % Cin) + half * 16;
        u16* dst = &As[buf][mloc * 32 + half * 16];
        if (ih >= 0 && ih < Hin && iw >= 0 && iw < Win) {
            const u16* sp = act + ((((size_t)bimg * Hin + ih) * Win + iw) * Cin + ci);
            CP_B128(sp, dst);
            CP_B128(sp + 8, dst + 8);
        } else {
            uint4 z = {0, 0, 0, 0};
            *(uint4*)dst = z;
            *(uint4*)(dst + 8) = z;
        }
    };
    auto loadB = [&](int k0, int buf) {
#if ENABLE_TDM
        if (wv == 0)
            tdm_issue_2d_bf16(lds_off_of(&Bs[buf][0]), wt + ((size_t)n0 * K + k0),
                              (unsigned)K, (unsigned)(n0 + TN),
                              32u, (unsigned)TN, (unsigned)K);
#else
        for (int c = tid; c < TN * 4; c += 256) {
            int nn = c >> 2, kk = (c & 3) * 8;
            CP_B128(&wt[(size_t)(n0 + nn) * K + k0 + kk], &Bs[buf][nn * 32 + kk]);
        }
#endif
    };

    loadA(0, 0);
    loadB(0, 0);
    constexpr int nk = K / 32;
    for (int i = 0; i < nk; i++) {
        int cur = i & 1;
#if HAVE_ASYNC
        WAIT_ASYNC();
#endif
#if ENABLE_TDM
        if (wv == 0) tdm_wait();
#endif
        __syncthreads();
        if (i + 1 < nk) { loadA((i + 1) * 32, cur ^ 1); loadB((i + 1) * 32, cur ^ 1); }
        Frag a;
        int mr = wv * 16 + ln;
        a.q[0] = *(const uint4*)&As[cur][mr * 32 + lh * 16];
        a.q[1] = *(const uint4*)&As[cur][mr * 32 + lh * 16 + 8];
#pragma unroll
        for (int t = 0; t < NT; t++) {
            Frag bb;
            int nc = t * 16 + ln;
            bb.q[0] = *(const uint4*)&Bs[cur][nc * 32 + lh * 16];
            bb.q[1] = *(const uint4*)&Bs[cur][nc * 32 + lh * 16 + 8];
            acc[t] = __builtin_amdgcn_wmma_f32_16x16x32_bf16(
                false, a.v, false, bb.v, (short)0, acc[t], false, false);
        }
        __syncthreads();
    }
    int mbase = blockIdx.x * 128 + wv * 16 + lh * 8;
#pragma unroll
    for (int t = 0; t < NT; t++) {
        int nc = n0 + t * 16 + ln;
        float bv = bias[nc];
        for (int r = 0; r < 8; r++) {
            float v = fmaxf(acc[t][r] + bv, 0.f);
            out[(size_t)(mbase + r) * Cout + nc] = f2bf(v);
        }
    }
}

// ---------------------------------------------------------------------------
// generic bf16 WMMA GEMM: C[m][n] = sum_k A[m][k] * B(k,n) (+bias, opt relu)
// TRANSB: B(k,n) = W[n*K+k] (rows padded to TN).  !TRANSB: B(k,n) = W[k*ldw+n].
// KFIX > 0: K is static and the whole TN x K weight panel is kept resident in
// LDS (loaded once via TDM).  A is double-buffered with async global->LDS.
// Writes f32 (Cf) and/or bf16 (Cb), row stride ldc; n bounds-checked vs Nreal.
// M % 128 == 0, K % 32 == 0.
// ---------------------------------------------------------------------------
template <int TN, bool TRANSB, int KFIX>
__global__ __launch_bounds__(256) void gemm_bf16(
    const u16* __restrict__ A, const u16* __restrict__ W,
    const float* __restrict__ bias, float* __restrict__ Cf, u16* __restrict__ Cb,
    int M, int Nreal, int Kdyn, int ldw, int ldc, int relu) {
    constexpr int NT = TN / 16;
    constexpr bool WHOLEB = (KFIX > 0);
    constexpr int BSZ = WHOLEB ? (TN * KFIX) : (2 * TN * 32);
    __shared__ u16 As[2][128 * 32];
    __shared__ u16 Bsh[BSZ];
    const int K = WHOLEB ? KFIX : Kdyn;
    int tid = threadIdx.x;
    int wv = tid >> 5, lane = tid & 31;
    int lh = lane >> 4, ln = lane & 15;
    int n0 = blockIdx.y * TN;
    int mloc = tid >> 1, half = tid & 1;
    size_t mrow = (size_t)blockIdx.x * 128 + mloc;

    v8f acc[NT];
    for (int i = 0; i < NT; i++)
        for (int r = 0; r < 8; r++) acc[i][r] = 0.f;

    auto loadA = [&](int k0, int buf) {
        const u16* ap = A + mrow * K + k0 + half * 16;
        u16* dst = &As[buf][mloc * 32 + half * 16];
        CP_B128(ap, dst);
        CP_B128(ap + 8, dst + 8);
    };
    auto loadB = [&](int k0, int buf) {
        u16* bb = &Bsh[buf * TN * 32];
        if (TRANSB) {
            for (int c = tid; c < TN * 4; c += 256) {
                int nn = c >> 2, kk = (c & 3) * 8;
                CP_B128(&W[(size_t)(n0 + nn) * K + k0 + kk], &bb[nn * 32 + kk]);
            }
        } else {
            for (int c = tid; c < TN * 32; c += 256) {
                int nn = c >> 5, kk = c & 31;
                bb[nn * 32 + kk] = W[(size_t)(k0 + kk) * ldw + n0 + nn];
            }
        }
    };

    if (WHOLEB) {
#if ENABLE_TDM
        if (wv == 0) {
            tdm_issue_2d_bf16(lds_off_of(&Bsh[0]), W + (size_t)n0 * K,
                              (unsigned)K, (unsigned)(n0 + TN),
                              (unsigned)K, (unsigned)TN, (unsigned)K);
            tdm_wait();
        }
#else
        for (int c = tid; c < TN * K / 8; c += 256) {
            int nn = c / (K / 8), kk = (c % (K / 8)) * 8;
            CP_B128(&W[(size_t)(n0 + nn) * K + kk], &Bsh[nn * K + kk]);
        }
#endif
    }
    loadA(0, 0);
    if (!WHOLEB) loadB(0, 0);
    int nk = K / 32;
    for (int i = 0; i < nk; i++) {
        int cur = i & 1;
#if HAVE_ASYNC
        WAIT_ASYNC();
#endif
        __syncthreads();
        if (i + 1 < nk) {
            loadA((i + 1) * 32, cur ^ 1);
            if (!WHOLEB) loadB((i + 1) * 32, cur ^ 1);
        }
        Frag a;
        int mr = wv * 16 + ln;
        a.q[0] = *(const uint4*)&As[cur][mr * 32 + lh * 16];
        a.q[1] = *(const uint4*)&As[cur][mr * 32 + lh * 16 + 8];
#pragma unroll
        for (int t = 0; t < NT; t++) {
            Frag bb;
            int nc = t * 16 + ln;
            const u16* bp = WHOLEB ? &Bsh[nc * KFIX + i * 32 + lh * 16]
                                   : &Bsh[cur * TN * 32 + nc * 32 + lh * 16];
            bb.q[0] = *(const uint4*)bp;
            bb.q[1] = *(const uint4*)(bp + 8);
            acc[t] = __builtin_amdgcn_wmma_f32_16x16x32_bf16(
                false, a.v, false, bb.v, (short)0, acc[t], false, false);
        }
        __syncthreads();
    }
    int mbase = blockIdx.x * 128 + wv * 16 + lh * 8;
#pragma unroll
    for (int t = 0; t < NT; t++) {
        int nc = n0 + t * 16 + ln;
        if (nc < Nreal) {
            float bv = bias ? bias[nc] : 0.f;
            for (int r = 0; r < 8; r++) {
                float v = acc[t][r] + bv;
                if (relu) v = fmaxf(v, 0.f);
                size_t oi = (size_t)(mbase + r) * ldc + nc;
                if (Cf) Cf[oi] = v;
                if (Cb) Cb[oi] = f2bf(v);
            }
        }
    }
    (void)M;
}

// ---------------------------------------------------------------------------
// SoftSOM: one block per (b, n); k = threadIdx.x in [0,256)
// ---------------------------------------------------------------------------
__global__ void som_kernel(const float* __restrict__ dotP, const float* __restrict__ dotG,
                           const float* __restrict__ znorm, const float* __restrict__ pn,
                           const float* __restrict__ gn, const float* __restrict__ temp_raw,
                           const float* __restrict__ gate_logits, u16* __restrict__ w_soft,
                           int Bn, int Kk) {
    int n = blockIdx.y, brow = blockIdx.x, k = threadIdx.x;
    __shared__ float red[256];
    size_t base = ((size_t)n * Bn + brow) * Kk;
    float zz = znorm[n * Bn + brow];
    float dp = dotP[base + k], dg = dotG[base + k];
    float d = sqrtf(fmaxf(zz + pn[n * Kk + k] - 2.f * dp, 0.f)) +
              sqrtf(fmaxf(zz + gn[n * Kk + k] - 2.f * dg, 0.f));
    float t = 1.f / (1.f + __expf(-temp_raw[n])) * 0.999f + 0.001f;
    float logit = -d / t;
    red[k] = logit; __syncthreads();
    for (int s = 128; s > 0; s >>= 1) {
        if (k < s) red[k] = fmaxf(red[k], red[k + s]);
        __syncthreads();
    }
    float mx = red[0]; __syncthreads();
    float e = __expf(logit - mx);
    red[k] = e; __syncthreads();
    for (int s = 128; s > 0; s >>= 1) {
        if (k < s) red[k] += red[k + s];
        __syncthreads();
    }
    float se = red[0]; __syncthreads();
    float g = 1.f / (1.f + __expf(-gate_logits[n * Kk + k]));
    float wv = (e / se) * g;
    red[k] = wv; __syncthreads();
    for (int s = 128; s > 0; s >>= 1) {
        if (k < s) red[k] += red[k + s];
        __syncthreads();
    }
    float sw = red[0];
    w_soft[base + k] = f2bf(wv / (sw + 1e-8f));
}

// ---------------------------------------------------------------------------
// tiny attention head (1 block): folds attention weights into the classifier:
// eff_w[c][n*256+d] = sum_h cls_w[c][h*256+d] * attnw[h][n]   (rows 10..15 = 0)
// ---------------------------------------------------------------------------
__global__ void attn_kernel(const float* __restrict__ class_emb, const float* __restrict__ q_w,
                            const float* __restrict__ q_b, const float* __restrict__ node_emb,
                            const float* __restrict__ attn_w, const float* __restrict__ attn_b,
                            const float* __restrict__ cls_w, u16* __restrict__ eff_w) {
    __shared__ float mean[256], q[256], part[256], aw[4][4];
    int t = threadIdx.x;
    float m = 0.f;
    for (int c = 0; c < 10; c++) m += class_emb[c * 256 + t];
    mean[t] = m * 0.1f;
    __syncthreads();
    float s = 0.f;
    for (int dd = 0; dd < 256; dd++) s += mean[dd] * q_w[t * 256 + dd];
    q[t] = s + q_b[t];
    __syncthreads();
    int p = t >> 4, chunk = t & 15;
    int h = p >> 2, n = p & 3;
    float acc = 0.f;
    for (int d = chunk * 16; d < chunk * 16 + 16; d++) {
        int row = h * 256 + d;
        float proj = attn_b[row];
        for (int dd = 0; dd < 256; dd++)
            proj += node_emb[n * 256 + dd] * attn_w[(size_t)row * 256 + dd];
        acc += q[d] * proj;
    }
    part[t] = acc;
    __syncthreads();
    if (chunk == 0) {
        float l = 0.f;
        for (int i = 0; i < 16; i++) l += part[p * 16 + i];
        aw[h][n] = l;
    }
    __syncthreads();
    if (t < 4) {
        float mx = -1e30f;
        for (int i = 0; i < 4; i++) mx = fmaxf(mx, aw[t][i]);
        float e[4], se = 0.f;
        for (int i = 0; i < 4; i++) { e[i] = __expf(aw[t][i] - mx); se += e[i]; }
        for (int i = 0; i < 4; i++) aw[t][i] = e[i] / se;
    }
    __syncthreads();
    for (int idx = t; idx < 16 * 1024; idx += 256) {
        int c = idx >> 10, r = idx & 1023;
        int n2 = r >> 8, d = r & 255;
        float v = 0.f;
        if (c < 10)
            for (int hh = 0; hh < 4; hh++)
                v += cls_w[c * 1024 + hh * 256 + d] * aw[hh][n2];
        eff_w[idx] = f2bf(v);
    }
}

// ---------------------------------------------------------------------------
// host driver
// ---------------------------------------------------------------------------
extern "C" void kernel_launch(void* const* d_in, const int* in_sizes, int n_in,
                              void* d_out, int out_size, void* d_ws, size_t ws_size,
                              hipStream_t stream) {
    (void)in_sizes; (void)n_in; (void)out_size; (void)ws_size;
    const float* x       = (const float*)d_in[0];
    const float* w1      = (const float*)d_in[1];
    const float* b1      = (const float*)d_in[2];
    const float* w2      = (const float*)d_in[3];
    const float* b2      = (const float*)d_in[4];
    const float* w3      = (const float*)d_in[5];
    const float* b3      = (const float*)d_in[6];
    const float* w4      = (const float*)d_in[7];
    const float* b4      = (const float*)d_in[8];
    const float* enc_w   = (const float*)d_in[9];
    const float* enc_b   = (const float*)d_in[10];
    const float* node_w  = (const float*)d_in[11];
    const float* node_b  = (const float*)d_in[12];
    const float* protos  = (const float*)d_in[13];
    const float* grid_p  = (const float*)d_in[14];
    const float* temp_raw    = (const float*)d_in[15];
    const float* gate_logits = (const float*)d_in[16];
    const float* class_emb   = (const float*)d_in[17];
    const float* node_emb    = (const float*)d_in[18];
    const float* q_w     = (const float*)d_in[19];
    const float* q_b     = (const float*)d_in[20];
    const float* attn_w  = (const float*)d_in[21];
    const float* attn_b  = (const float*)d_in[22];
    const float* cls_w   = (const float*)d_in[23];
    const float* cls_b   = (const float*)d_in[24];
    float* out = (float*)d_out;

    const int B = 4096, D = 256, N = 4, K = 256;

    char* wsb = (char*)d_ws;
    size_t off = 0;
    auto alloc = [&](size_t bytes) -> void* {
        void* p = wsb + off;
        off = (off + bytes + 255) & ~(size_t)255;
        return p;
    };
    u16*   h1      = (u16*)alloc((size_t)B * 15 * 15 * 32 * 2);
    u16*   h2      = (u16*)alloc((size_t)B * 64 * 64 * 2);
    u16*   h3      = (u16*)alloc((size_t)B * 64 * 128 * 2);
    u16*   h4      = (u16*)alloc((size_t)B * 64 * 256 * 2);
    u16*   w2t     = (u16*)alloc(64 * 288 * 2);
    u16*   w3t     = (u16*)alloc(128 * 576 * 2);
    u16*   w4t     = (u16*)alloc(256 * 1152 * 2);
    u16*   enc_wt  = (u16*)alloc((size_t)256 * 16384 * 2);
    u16*   node_wb = (u16*)alloc((size_t)N * D * D * 2);
    u16*   protos_b= (u16*)alloc((size_t)N * K * D * 2);
    u16*   grid_b  = (u16*)alloc((size_t)N * K * D * 2);
    u16*   z0b     = (u16*)alloc((size_t)B * D * 2);
    float* zf      = (float*)alloc((size_t)N * B * D * 4);
    u16*   zb      = (u16*)alloc((size_t)N * B * D * 2);
    float* znorm   = (float*)alloc((size_t)N * B * 4);
    float* pn      = (float*)alloc((size_t)N * K * 4);
    float* gn      = (float*)alloc((size_t)N * K * 4);
    float* dotP    = (float*)alloc((size_t)N * B * K * 4);
    float* dotG    = (float*)alloc((size_t)N * B * K * 4);
    u16*   wsoft   = (u16*)alloc((size_t)N * B * K * 2);
    u16*   topo    = (u16*)alloc((size_t)B * N * D * 2);
    u16*   effw    = (u16*)alloc(16 * 1024 * 2);

    auto nb = [](size_t total) { return (unsigned)((total + 255) / 256); };

    // weight conversions / permutations
    cvt_conv_w<<<nb(64 * 288), 256, 0, stream>>>(w2, w2t, 64, 32, 3);
    cvt_conv_w<<<nb(128 * 576), 256, 0, stream>>>(w3, w3t, 128, 64, 3);
    cvt_conv_w<<<nb(256 * 1152), 256, 0, stream>>>(w4, w4t, 256, 128, 3);
    cvt_enc_w<<<nb((size_t)256 * 16384), 256, 0, stream>>>(enc_w, enc_wt);
    cvt_bf_kernel<<<nb((size_t)N * D * D), 256, 0, stream>>>(node_w, node_wb, (size_t)N * D * D);
    cvt_bf_kernel<<<nb((size_t)N * K * D), 256, 0, stream>>>(protos, protos_b, (size_t)N * K * D);
    cvt_bf_kernel<<<nb((size_t)N * K * D), 256, 0, stream>>>(grid_p, grid_b, (size_t)N * K * D);
    row_sqnorm<<<nb(N * K), 256, 0, stream>>>(protos, pn, N * K, D);
    row_sqnorm<<<nb(N * K), 256, 0, stream>>>(grid_p, gn, N * K, D);

    // encoder
    conv1_kernel<<<nb((size_t)B * 15 * 15 * 32), 256, 0, stream>>>(x, w1, b1, h1);
    conv_igemm<32, 3, 2, 1, 15, 15, 8, 8, 64, 64>
        <<<dim3(B * 64 / 128, 1), 256, 0, stream>>>(h1, w2t, b2, h2);
    conv_igemm<64, 3, 1, 1, 8, 8, 8, 8, 128, 128>
        <<<dim3(B * 64 / 128, 1), 256, 0, stream>>>(h2, w3t, b3, h3);
    conv_igemm<128, 3, 1, 1, 8, 8, 8, 8, 256, 128>
        <<<dim3(B * 64 / 128, 2), 256, 0, stream>>>(h3, w4t, b4, h4);
    gemm_bf16<128, true, 0><<<dim3(B / 128, 2), 256, 0, stream>>>(
        h4, enc_wt, enc_b, (float*)nullptr, z0b, B, D, 16384, 0, D, 0);

    // per-node fc + distance dot products (whole weight panel resident in LDS)
    for (int n = 0; n < N; n++) {
        gemm_bf16<128, true, 256><<<dim3(B / 128, 2), 256, 0, stream>>>(
            z0b, node_wb + (size_t)n * D * D, node_b + n * D,
            zf + (size_t)n * B * D, zb + (size_t)n * B * D, B, D, D, 0, D, 0);
    }
    row_sqnorm<<<nb(N * B), 256, 0, stream>>>(zf, znorm, N * B, D);
    for (int n = 0; n < N; n++) {
        gemm_bf16<128, true, 256><<<dim3(B / 128, 2), 256, 0, stream>>>(
            zb + (size_t)n * B * D, protos_b + (size_t)n * K * D, (const float*)nullptr,
            dotP + (size_t)n * B * K, (u16*)nullptr, B, K, D, 0, K, 0);
        gemm_bf16<128, true, 256><<<dim3(B / 128, 2), 256, 0, stream>>>(
            zb + (size_t)n * B * D, grid_b + (size_t)n * K * D, (const float*)nullptr,
            dotG + (size_t)n * B * K, (u16*)nullptr, B, K, D, 0, K, 0);
    }

    // SoftSOM weights
    som_kernel<<<dim3(B, N), 256, 0, stream>>>(dotP, dotG, znorm, pn, gn,
                                               temp_raw, gate_logits, wsoft, B, K);

    // topo: w_soft[n] @ protos[n]  ->  written interleaved as [B][n*D+d]
    for (int n = 0; n < N; n++) {
        gemm_bf16<128, false, 0><<<dim3(B / 128, 2), 256, 0, stream>>>(
            wsoft + (size_t)n * B * K, protos_b + (size_t)n * K * D, (const float*)nullptr,
            (float*)nullptr, topo + n * D, B, D, K, D, N * D, 0);
    }

    // attention folded into classifier weights, then final WMMA GEMM
    attn_kernel<<<1, 256, 0, stream>>>(class_emb, q_w, q_b, node_emb,
                                       attn_w, attn_b, cls_w, effw);
    gemm_bf16<16, true, 1024><<<dim3(B / 128, 1), 256, 0, stream>>>(
        topo, effw, cls_b, out, (u16*)nullptr, B, 10, N * D, 0, 10, 0);
}
